// GraphSim_21680994910658
// MI455X (gfx1250) — matrix operand order repaired and checked by reading, not compiled, not run on hardware
//
#include <hip/hip_runtime.h>

typedef __attribute__((ext_vector_type(16))) _Float16 v16h;
typedef __attribute__((ext_vector_type(8)))  _Float16 v8h;
typedef __attribute__((ext_vector_type(8)))  float    v8f;

#define NNODES 32768   // B*N
#define NPAIR  512
#define NPN    64
#define NEDGE  262144  // B*E

// ---------------- degree / dinv ----------------
__global__ void k_set1(float* p, int n) {
  int i = blockIdx.x * blockDim.x + threadIdx.x;
  if (i < n) p[i] = 1.0f;
}

__global__ void k_scatter_deg(const int* __restrict__ dst, float* deg, int ne) {
  int i = blockIdx.x * blockDim.x + threadIdx.x;
  if (i < ne) atomicAdd(&deg[dst[i]], 1.0f);
}

__global__ void k_rsqrt(float* p, int n) {
  int i = blockIdx.x * blockDim.x + threadIdx.x;
  if (i < n) p[i] = rsqrtf(p[i]);
}

// ---------------- f32 -> f16 conversion (A operands), zero padded K->Kpad ----
__global__ void k_cvt(const float* __restrict__ in, _Float16* __restrict__ out,
                      int M, int K, int Kpad, int relu) {
  int t = blockIdx.x * blockDim.x + threadIdx.x;
  if (t >= M * Kpad) return;
  int m = t / Kpad, k = t - m * Kpad;
  float v = (k < K) ? in[(size_t)m * K + k] : 0.0f;
  if (relu) v = fmaxf(v, 0.0f);
  out[t] = (_Float16)v;
}

// ---- weight f32[K][N] -> f16 transposed Bt[N][Kpad], zero padded ----
__global__ void k_cvt_wt(const float* __restrict__ in, _Float16* __restrict__ out,
                         int K, int N, int Kpad) {
  int t = blockIdx.x * blockDim.x + threadIdx.x;
  if (t >= N * Kpad) return;
  int n = t / Kpad, k = t - n * Kpad;
  out[t] = (_Float16)((k < K) ? in[(size_t)k * N + n] : 0.0f);
}

// ---------------- GCN aggregate ----------------
// out[i,f] = h[i,f]*dinv[i]^2 + bias[f]
__global__ void k_self(const float* __restrict__ h, const float* __restrict__ dinv,
                       const float* __restrict__ bias, float* __restrict__ out,
                       int nrows, int F) {
  int t = blockIdx.x * blockDim.x + threadIdx.x;
  if (t >= nrows * F) return;
  int i = t / F, f = t - i * F;
  float dv = dinv[i];
  out[t] = h[t] * dv * dv + bias[f];
}

// out[dst,f] += h[src,f] * dinv[src]*dinv[dst]
__global__ void k_edge(const float* __restrict__ h, const int* __restrict__ src,
                       const int* __restrict__ dst, const float* __restrict__ dinv,
                       float* out, int ne, int F) {
  int t = blockIdx.x * blockDim.x + threadIdx.x;
  if (t >= ne * F) return;
  int e = t / F, f = t - e * F;
  int s = src[e], d = dst[e];
  atomicAdd(&out[d * F + f], h[s * F + f] * dinv[s] * dinv[d]);
}

// ---- WMMA GEMM: C[M,N] = A[M,Kpad](f16) * Bt[N,Kpad]^T(f16) ----
// One 16x(16*NT) strip per wave: A fragment loaded once, reused across NT WMMAs.
template<int NT, bool RELU_OUT, bool HAS_BIAS>
__global__ void k_gemm(const _Float16* __restrict__ A, const _Float16* __restrict__ Bt,
                       const float* __restrict__ bias, float* __restrict__ C,
                       int M, int N, int Kpad) {
  int ntn = N / (16 * NT);
  int ntiles = (M >> 4) * ntn;
  int tile = blockIdx.x * (blockDim.x >> 5) + (threadIdx.x >> 5);
  if (tile >= ntiles) return;
  int tm = (tile / ntn) << 4;
  int tn = (tile % ntn) * (16 * NT);
  int lane = threadIdx.x & 31;
  int hs = lane >> 4, r = lane & 15;
  // A fragment: lane-half hs covers K = k0+hs*16 .. +15 (contiguous)
  const _Float16* Arow = A + (size_t)(tm + r) * Kpad + hs * 16;
  // B fragments: lane-half hs covers K = k0+hs*16 .. +15 (contiguous in Bt row)
  size_t bbase = (size_t)(tn + r) * Kpad + hs * 16;
  v8f acc[NT] = {};
  for (int k0 = 0; k0 < Kpad; k0 += 32) {
    v8h alo = *(const v8h*)(Arow + k0);
    v8h ahi = *(const v8h*)(Arow + k0 + 8);
    v16h a = __builtin_shufflevector(alo, ahi, 0, 1, 2, 3, 4, 5, 6, 7,
                                     8, 9, 10, 11, 12, 13, 14, 15);
#pragma unroll
    for (int t = 0; t < NT; ++t) {
      v16h b = *(const v16h*)(Bt + bbase + (size_t)(16 * t) * Kpad + k0);
      acc[t] = __builtin_amdgcn_wmma_f32_16x16x32_f16(false, a, false, b,
                                                      (short)0, acc[t], false, false);
    }
  }
#pragma unroll
  for (int t = 0; t < NT; ++t) {
    int nc = tn + 16 * t + r;
#pragma unroll
    for (int rr = 0; rr < 8; ++rr) {
      int m = tm + hs * 8 + rr;
      float v = acc[t][rr];
      if (HAS_BIAS) v += bias[nc];
      if (RELU_OUT) v = fmaxf(v, 0.0f);
      C[(size_t)m * N + nc] = v;
    }
  }
}

template<bool RELU_OUT, bool HAS_BIAS>
static void launch_gemm(const _Float16* A, const _Float16* Bt, const float* bias,
                        float* C, int M, int N, int Kpad, hipStream_t stream) {
  if (N % 64 == 0) {
    int ntiles = (M / 16) * (N / 64);
    k_gemm<4, RELU_OUT, HAS_BIAS><<<(ntiles + 3) / 4, 128, 0, stream>>>(A, Bt, bias, C, M, N, Kpad);
  } else if (N % 32 == 0) {
    int ntiles = (M / 16) * (N / 32);
    k_gemm<2, RELU_OUT, HAS_BIAS><<<(ntiles + 3) / 4, 128, 0, stream>>>(A, Bt, bias, C, M, N, Kpad);
  } else {
    int ntiles = (M / 16) * (N / 16);
    k_gemm<1, RELU_OUT, HAS_BIAS><<<(ntiles + 3) / 4, 128, 0, stream>>>(A, Bt, bias, C, M, N, Kpad);
  }
}

// ---- similarity: S[b] = Q_b (64xKpad) * C_b^T, one 16x64 strip per wave ----
__global__ void k_sim(const _Float16* __restrict__ Q16, const _Float16* __restrict__ C16,
                      float* __restrict__ S, int Kpad) {
  int blk = blockIdx.x * (blockDim.x >> 5) + (threadIdx.x >> 5);
  if (blk >= NPAIR * 4) return;
  int b = blk >> 2;
  int tm = (blk & 3) << 4;
  int lane = threadIdx.x & 31, hs = lane >> 4, r = lane & 15;
  const _Float16* Arow = Q16 + (size_t)(b * NPN + tm + r) * Kpad + hs * 16;
  size_t bbase = (size_t)(b * NPN + r) * Kpad + hs * 16;
  v8f acc[4] = {};
  for (int k0 = 0; k0 < Kpad; k0 += 32) {
    v8h alo = *(const v8h*)(Arow + k0);
    v8h ahi = *(const v8h*)(Arow + k0 + 8);
    v16h a = __builtin_shufflevector(alo, ahi, 0, 1, 2, 3, 4, 5, 6, 7,
                                     8, 9, 10, 11, 12, 13, 14, 15);
#pragma unroll
    for (int t = 0; t < 4; ++t) {
      v16h bb = *(const v16h*)(C16 + bbase + (size_t)(16 * t) * Kpad + k0);
      acc[t] = __builtin_amdgcn_wmma_f32_16x16x32_f16(false, a, false, bb,
                                                      (short)0, acc[t], false, false);
    }
  }
#pragma unroll
  for (int t = 0; t < 4; ++t)
#pragma unroll
    for (int rr = 0; rr < 8; ++rr)
      S[((size_t)b * NPN + tm + hs * 8 + rr) * NPN + 16 * t + r] = acc[t][rr];
}

// ---------------- fused 5x5 SAME conv + ReLU + 2x2 maxpool ----------------
__global__ void k_conv_pool(const float* __restrict__ in, const float* __restrict__ w,
                            const float* __restrict__ bias, float* __restrict__ out,
                            int Bn, int Cin, int Cout, int Hin, int outCtot, int coOff) {
  int Ho = Hin >> 1;
  int total = Bn * Cout * Ho * Ho;
  int t = blockIdx.x * blockDim.x + threadIdx.x;
  if (t >= total) return;
  int px = t % Ho; int u = t / Ho;
  int py = u % Ho; u /= Ho;
  int co = u % Cout; int b = u / Cout;
  const float* wc = w + (size_t)co * Cin * 25;
  float bv = bias[co];
  float best = -3.4e38f;
  for (int dy = 0; dy < 2; ++dy)
    for (int dx = 0; dx < 2; ++dx) {
      int iy = py * 2 + dy, ix = px * 2 + dx;
      float sum = bv;
      for (int ci = 0; ci < Cin; ++ci) {
        const float* ip = in + ((size_t)(b * Cin + ci) * Hin) * Hin;
        const float* wp = wc + ci * 25;
        for (int ky = 0; ky < 5; ++ky) {
          int yy = iy + ky - 2;
          if (yy < 0 || yy >= Hin) continue;
          for (int kx = 0; kx < 5; ++kx) {
            int xx = ix + kx - 2;
            if (xx < 0 || xx >= Hin) continue;
            sum += ip[yy * Hin + xx] * wp[ky * 5 + kx];
          }
        }
      }
      sum = fmaxf(sum, 0.0f);
      best = fmaxf(best, sum);
    }
  out[(((size_t)b * outCtot + coOff + co) * Ho + py) * Ho + px] = best;
}

// ---------------- final score ----------------
__global__ void k_score(const float* __restrict__ h2, const float* __restrict__ sw,
                        const float* __restrict__ sb, float* __restrict__ out, int Bn) {
  int b = blockIdx.x * blockDim.x + threadIdx.x;
  if (b >= Bn) return;
  float s = sb[0];
#pragma unroll
  for (int j = 0; j < 16; ++j) s += h2[b * 16 + j] * sw[j];
  out[b] = s;
}

static inline int up256(size_t n) { return (int)((n + 255) / 256); }

extern "C" void kernel_launch(void* const* d_in, const int* in_sizes, int n_in,
                              void* d_out, int out_size, void* d_ws, size_t ws_size,
                              hipStream_t stream) {
  const float* x_q = (const float*)d_in[0];
  const float* x_c = (const float*)d_in[1];
  const int*   eq  = (const int*)d_in[2];
  const int*   ec  = (const int*)d_in[3];
  const float* gw[3] = {(const float*)d_in[4], (const float*)d_in[6], (const float*)d_in[8]};
  const float* gb[3] = {(const float*)d_in[5], (const float*)d_in[7], (const float*)d_in[9]};
  const float* cw[3] = {(const float*)d_in[10], (const float*)d_in[12], (const float*)d_in[14]};
  const float* cb[3] = {(const float*)d_in[11], (const float*)d_in[13], (const float*)d_in[15]};
  const float* lw1 = (const float*)d_in[16];
  const float* lb1 = (const float*)d_in[17];
  const float* lw2 = (const float*)d_in[18];
  const float* lb2 = (const float*)d_in[19];
  const float* sw  = (const float*)d_in[20];
  const float* sb  = (const float*)d_in[21];
  float* out = (float*)d_out;

  // ---- workspace carve-out: f32 region then f16 region ----
  float* Wf = (float*)d_ws;
  size_t offf = 0;
  auto af = [&](size_t n) { float* p = Wf + offf; offf += n; return p; };
  float* dinv[2] = {af(NNODES), af(NNODES)};
  float* qf[3]  = {af((size_t)NNODES * 64), af((size_t)NNODES * 32), af((size_t)NNODES * 16)};
  float* cfe[3] = {af((size_t)NNODES * 64), af((size_t)NNODES * 32), af((size_t)NNODES * 16)};
  float* tmp  = af((size_t)NNODES * 64);
  float* sim  = af((size_t)NPAIR * 64 * 64);
  float* p1   = af((size_t)NPAIR * 8 * 32 * 32);
  float* p2   = af((size_t)NPAIR * 16 * 16 * 16);
  float* feat = af((size_t)NPAIR * 6144);
  float* h1   = af((size_t)NPAIR * 256);
  float* h2   = af((size_t)NPAIR * 16);

  _Float16* Wh = (_Float16*)(Wf + offf);
  size_t offh = 0;
  auto ah = [&](size_t n) { _Float16* p = Wh + offh; offh += n; return p; };
  _Float16* gw16[3] = {ah(64 * 32), ah(32 * 64), ah(16 * 32)};  // Bt[N][Kpad]
  _Float16* x16   = ah((size_t)NNODES * 32);                    // layer-0 A (reused q/c)
  _Float16* act16 = ah((size_t)NNODES * 64);                    // relu'd A (reused)
  _Float16* qf16[3] = {ah((size_t)NNODES * 64), ah((size_t)NNODES * 32), ah((size_t)NNODES * 32)};
  _Float16* cf16[3] = {ah((size_t)NNODES * 64), ah((size_t)NNODES * 32), ah((size_t)NNODES * 32)};
  _Float16* feat16 = ah((size_t)NPAIR * 6144);
  _Float16* h116   = ah((size_t)NPAIR * 256);
  _Float16* wbuf   = ah((size_t)256 * 6144);                    // MLP Bt (reused)

  const int dims[4]  = {16, 64, 32, 16};
  const int kpadL[4] = {32, 64, 32, 32};  // dims rounded up to 32

  // GCN weights -> transposed f16 (once, reused for both graphs)
  for (int l = 0; l < 3; ++l)
    k_cvt_wt<<<up256((size_t)dims[l + 1] * kpadL[l]), 256, 0, stream>>>(
        gw[l], gw16[l], dims[l], dims[l + 1], kpadL[l]);

  // -------- GCN passes (q then c) --------
  for (int g = 0; g < 2; ++g) {
    const float* x = g ? x_c : x_q;
    const int* src = g ? ec : eq;
    const int* dst = src + NEDGE;
    float* dv = dinv[g];
    float** fb = g ? cfe : qf;
    _Float16** fb16 = g ? cf16 : qf16;

    k_set1<<<up256(NNODES), 256, 0, stream>>>(dv, NNODES);
    k_scatter_deg<<<up256(NEDGE), 256, 0, stream>>>(dst, dv, NEDGE);
    k_rsqrt<<<up256(NNODES), 256, 0, stream>>>(dv, NNODES);

    k_cvt<<<up256((size_t)NNODES * 32), 256, 0, stream>>>(x, x16, NNODES, 16, 32, 0);

    for (int l = 0; l < 3; ++l) {
      int Nd = dims[l + 1], Kp = kpadL[l];
      const _Float16* Ain = (l == 0) ? x16 : act16;
      launch_gemm<false, false>(Ain, gw16[l], nullptr, tmp, NNODES, Nd, Kp, stream);
      k_self<<<up256((size_t)NNODES * Nd), 256, 0, stream>>>(tmp, dv, gb[l], fb[l], NNODES, Nd);
      k_edge<<<up256((size_t)NEDGE * Nd), 256, 0, stream>>>(tmp, src, dst, dv, fb[l], NEDGE, Nd);
      // pre-ReLU f16 copy for similarity
      k_cvt<<<up256((size_t)NNODES * kpadL[l + 1]), 256, 0, stream>>>(
          fb[l], fb16[l], NNODES, Nd, kpadL[l + 1], 0);
      // ReLU'd f16 copy as next layer's A operand
      if (l < 2)
        k_cvt<<<up256((size_t)NNODES * kpadL[l + 1]), 256, 0, stream>>>(
            fb[l], act16, NNODES, Nd, kpadL[l + 1], 1);
    }
  }

  // -------- similarity + conv towers --------
  for (int m = 0; m < 3; ++m) {
    k_sim<<<(NPAIR * 4 + 3) / 4, 128, 0, stream>>>(qf16[m], cf16[m], sim, kpadL[m + 1]);
    k_conv_pool<<<up256((size_t)NPAIR * 8 * 32 * 32), 256, 0, stream>>>(
        sim, cw[0] + m * 8 * 1 * 25, cb[0] + m * 8, p1, NPAIR, 1, 8, 64, 8, 0);
    k_conv_pool<<<up256((size_t)NPAIR * 16 * 16 * 16), 256, 0, stream>>>(
        p1, cw[1] + m * 16 * 8 * 25, cb[1] + m * 16, p2, NPAIR, 8, 16, 32, 16, 0);
    k_conv_pool<<<up256((size_t)NPAIR * 32 * 8 * 8), 256, 0, stream>>>(
        p2, cw[2] + m * 32 * 16 * 25, cb[2] + m * 32, feat, NPAIR, 16, 32, 16, 96, m * 32);
  }

  // -------- MLP head --------
  {
    k_cvt<<<up256((size_t)NPAIR * 6144), 256, 0, stream>>>(feat, feat16, NPAIR, 6144, 6144, 0);
    k_cvt_wt<<<up256((size_t)256 * 6144), 256, 0, stream>>>(lw1, wbuf, 6144, 256, 6144);
    launch_gemm<true, true>(feat16, wbuf, lb1, h1, NPAIR, 256, 6144, stream);

    k_cvt<<<up256((size_t)NPAIR * 256), 256, 0, stream>>>(h1, h116, NPAIR, 256, 256, 0);
    k_cvt_wt<<<up256((size_t)16 * 256), 256, 0, stream>>>(lw2, wbuf, 256, 16, 256);
    launch_gemm<true, true>(h116, wbuf, lb2, h2, NPAIR, 16, 256, stream);

    k_score<<<up256(NPAIR), 256, 0, stream>>>(h2, sw, sb, out, NPAIR);
  }
}
